// GPAM_18743237280293
// MI455X (gfx1250) — matrix operand order repaired and checked by reading, not compiled, not run on hardware
//
#include <hip/hip_runtime.h>
#include <hip/hip_bf16.h>
#include <math.h>

// ---------------------------------------------------------------------------
// Exact simplification: the reference LayerNorms over a size-1 axis, so the
// input tensor collapses to the scalar ln_in_b and the output is batch-
// independent. We run the 6-layer transformer on one (500 x 512) sequence and
// broadcast the final 256-class softmax to all 32 batch rows.
//
// All GEMMs use v_wmma_f32_16x16x32_f16 (wave32, f16 in / f32 acc).
// Weight-side (B) operands are pre-packed once per call into f16 in exact
// WMMA fragment order (zero-padded): one contiguous 32B load per lane per
// B tile, zero converts in the GEMM hot loop. The packed GEMM is software-
// pipelined (prefetch next A/B fragments under the current WMMAs) and GEMM
// kernels request waves_per_eu(1) for full VGPR budget (1-wave blocks).
// ---------------------------------------------------------------------------

typedef __attribute__((ext_vector_type(16))) _Float16 v16h;
typedef __attribute__((ext_vector_type(8)))  float    v8f;

#define N_TOK    500
#define DMODEL   512
#define KDIM     32
#define QKVW     160      // HS*KD = 5*32
#define NSCALE   3
#define NLAYER   6
#define FFDIM    2048
#define CONCAT_W 1536     // NS * D
#define NCLASS   256
#define NBATCH   32

__device__ __forceinline__ float gelu_exact(float x) {
  return 0.5f * x * (1.0f + erff(x * 0.7071067811865475f));
}
__device__ __forceinline__ int imin(int a, int b) { return a < b ? a : b; }

__device__ __forceinline__ v16h cvt_frag(float4 w0, float4 w1, float4 w2, float4 w3) {
  v16h r;
  r[0]  = (_Float16)w0.x; r[1]  = (_Float16)w0.y; r[2]  = (_Float16)w0.z; r[3]  = (_Float16)w0.w;
  r[4]  = (_Float16)w1.x; r[5]  = (_Float16)w1.y; r[6]  = (_Float16)w1.z; r[7]  = (_Float16)w1.w;
  r[8]  = (_Float16)w2.x; r[9]  = (_Float16)w2.y; r[10] = (_Float16)w2.z; r[11] = (_Float16)w2.w;
  r[12] = (_Float16)w3.x; r[13] = (_Float16)w3.y; r[14] = (_Float16)w3.z; r[15] = (_Float16)w3.w;
  return r;
}

// A fragment: elements 0-15 at p[0..15]
// (p = Arow + k0 + 16*half; contiguous k order). 16B-aligned.
__device__ __forceinline__ v16h load_afrag(const float* __restrict__ p) {
  float4 w0 = *reinterpret_cast<const float4*>(p);
  float4 w1 = *reinterpret_cast<const float4*>(p + 4);
  float4 w2 = *reinterpret_cast<const float4*>(p + 8);
  float4 w3 = *reinterpret_cast<const float4*>(p + 12);
  return cvt_frag(w0, w1, w2, w3);
}

// B fragment (on-the-fly transposed path): 16 contiguous floats at p.
__device__ __forceinline__ v16h load_bfrag16(const float* __restrict__ p) {
  float4 w0 = *reinterpret_cast<const float4*>(p);
  float4 w1 = *reinterpret_cast<const float4*>(p + 4);
  float4 w2 = *reinterpret_cast<const float4*>(p + 8);
  float4 w3 = *reinterpret_cast<const float4*>(p + 12);
  return cvt_frag(w0, w1, w2, w3);
}

// ---------------------------------------------------------------------------
// Pack B (K x N, f32 row-major) into WMMA fragment order, f16, zero-padded:
//   P[((tk*NTt + tn)*32 + lane)*16 + e] = B[(tk*32 + 16*(lane>>4) + e)][tn*16 + (lane&15)]
// ---------------------------------------------------------------------------
__global__ void pack_b_kernel(const float* __restrict__ B, int ldb, int K, int N,
                              int NTt, long total, _Float16* __restrict__ P) {
  long t = (long)blockIdx.x * blockDim.x + threadIdx.x;
  if (t >= total) return;
  int  lane = (int)(t & 31);
  long tile = t >> 5;
  int  tn = (int)(tile % NTt);
  int  tk = (int)(tile / NTt);
  int  half = lane >> 4, q = lane & 15;
  int  n = tn * 16 + q;
  v16h frag;
#pragma unroll
  for (int e = 0; e < 16; ++e) {
    int k = tk * 32 + half * 16 + e;
    float v = (k < K && n < N) ? B[(long)k * ldb + n] : 0.0f;
    frag[e] = (_Float16)v;
  }
  *reinterpret_cast<v16h*>(P + t * 16) = frag;
}

// ---------------------------------------------------------------------------
// Packed-B WMMA GEMM: C = act(alpha * A @ B + bias). A: f32 row-major.
// B: fragment-packed f16 (NTt total n-tiles, starting at tile ntBase).
// Software-pipelined: next K-step's fragments are fetched under the current
// WMMAs. No K guards (pack is zero-padded; A rows readable up to KT*32).
// Row/col overrun handled by address clamping into never-stored C rows/cols.
// ---------------------------------------------------------------------------
template<int NT>
__global__ void __attribute__((amdgpu_waves_per_eu(1)))
gemm_wmma_pk(const float* __restrict__ A, int lda,
             const _Float16* __restrict__ BP, int NTt, int ntBase,
             const float* __restrict__ bias,
             float* __restrict__ C, int ldc,
             int M, int N, int KT, float alpha, int act) {
  const int m0   = blockIdx.x << 4;
  const int n0   = (blockIdx.y * NT) << 4;
  const int lane = threadIdx.x;
  const int half = lane >> 4;
  const int q    = lane & 15;
  const int arow = imin(m0 + q, M - 1);
  const int lastTile = ((N + 15) >> 4) - 1;

  const v16h* bp[NT];
#pragma unroll
  for (int j = 0; j < NT; ++j) {
    int tn = ntBase + imin((n0 >> 4) + j, lastTile);
    bp[j] = reinterpret_cast<const v16h*>(BP) + (long)tn * 32 + lane;
  }
  const long bstep = (long)NTt * 32;   // v16h elements per K-step

  v8f acc[NT];
#pragma unroll
  for (int j = 0; j < NT; ++j) { v8f z = {}; acc[j] = z; }

  const float* ap = A + (long)arow * lda + (half << 4);

  // Prologue: fetch K-step 0.
  v16h a = load_afrag(ap);
  v16h b[NT];
#pragma unroll
  for (int j = 0; j < NT; ++j) b[j] = *bp[j];

  for (int tk = 0; tk < KT - 1; ++tk) {
    // Prefetch next K-step into fresh temporaries...
    v16h an = load_afrag(ap + ((tk + 1) << 5));
    v16h bn[NT];
#pragma unroll
    for (int j = 0; j < NT; ++j) { bp[j] += bstep; bn[j] = *bp[j]; }
    // ...while the current fragments feed the matrix pipe.
#pragma unroll
    for (int j = 0; j < NT; ++j)
      acc[j] = __builtin_amdgcn_wmma_f32_16x16x32_f16(false, a, false, b[j],
                                                      (short)0, acc[j], false, false);
    a = an;
#pragma unroll
    for (int j = 0; j < NT; ++j) b[j] = bn[j];
  }
#pragma unroll
  for (int j = 0; j < NT; ++j)
    acc[j] = __builtin_amdgcn_wmma_f32_16x16x32_f16(false, a, false, b[j],
                                                    (short)0, acc[j], false, false);

#pragma unroll
  for (int j = 0; j < NT; ++j) {
    int cn = n0 + (j << 4) + q;
    if (cn < N) {
      float bb = bias ? bias[cn] : 0.0f;
#pragma unroll
      for (int v = 0; v < 8; ++v) {   // C/D: row = v + 8*half, col = lane&15
        int cm = m0 + v + (half << 3);
        if (cm < M) {
          float x = acc[j][v] * alpha + bb;
          if (act) x = gelu_exact(x);
          C[(long)cm * ldc + cn] = x;
        }
      }
    }
  }
}

// ---------------------------------------------------------------------------
// QK^T GEMM (on-the-fly f16): C[m][n] = alpha * sum_k A[m][k] * B[n][k].
// K is a multiple of 32 (here K == 32). NT = 4 n-tiles per wave.
// ---------------------------------------------------------------------------
__global__ void __attribute__((amdgpu_waves_per_eu(1)))
gemm_wmma_qk(const float* __restrict__ A, int lda,
             const float* __restrict__ B, int ldb,
             float* __restrict__ C, int ldc,
             int M, int N, int K, float alpha) {
  const int NT = 4;
  const int m0   = blockIdx.x << 4;
  const int n0   = (blockIdx.y * NT) << 4;
  const int lane = threadIdx.x;
  const int half = lane >> 4;
  const int q    = lane & 15;
  const int arow = imin(m0 + q, M - 1);

  const float* brow[NT];
#pragma unroll
  for (int j = 0; j < NT; ++j) {
    int bcol = imin(n0 + (j << 4) + q, N - 1);
    brow[j] = B + (long)bcol * ldb + (half << 4);
  }

  v8f acc[NT];
#pragma unroll
  for (int j = 0; j < NT; ++j) { v8f z = {}; acc[j] = z; }

  const float* ap = A + (long)arow * lda + (half << 4);
  for (int k0 = 0; k0 < K; k0 += 32) {
    v16h a = load_afrag(ap + k0);
    v16h b[NT];
#pragma unroll
    for (int j = 0; j < NT; ++j) b[j] = load_bfrag16(brow[j] + k0);
#pragma unroll
    for (int j = 0; j < NT; ++j)
      acc[j] = __builtin_amdgcn_wmma_f32_16x16x32_f16(false, a, false, b[j],
                                                      (short)0, acc[j], false, false);
  }

#pragma unroll
  for (int j = 0; j < NT; ++j) {
    int cn = n0 + (j << 4) + q;
    if (cn < N) {
#pragma unroll
      for (int v = 0; v < 8; ++v) {
        int cm = m0 + v + (half << 3);
        if (cm < M) C[(long)cm * ldc + cn] = acc[j][v] * alpha;
      }
    }
  }
}

// ---------------------------------------------------------------------------
// Elementwise / reduction helpers
// ---------------------------------------------------------------------------
__global__ void colsum_kernel(const float* __restrict__ W, float* __restrict__ cs,
                              int P, int D) {
  int d = blockIdx.x * blockDim.x + threadIdx.x;
  if (d >= D) return;
  float s = 0.f;
  for (int p = 0; p < P; ++p) s += W[(long)p * D + d];
  cs[d] = s;
}

__global__ void h0_kernel(const float* __restrict__ cs, const float* __restrict__ pb,
                          const float* __restrict__ pos, const float* __restrict__ bscal,
                          float* __restrict__ h, int total, int D) {
  int i = blockIdx.x * blockDim.x + threadIdx.x;
  if (i >= total) return;
  int d = i % D;
  h[i] = bscal[0] * cs[d] + pb[d] + pos[i];
}

__global__ void pool_kernel(const float* __restrict__ h, float* __restrict__ pooled,
                            int s, int total, int D) {
  int i = blockIdx.x * blockDim.x + threadIdx.x;
  if (i >= total) return;
  int n = i / D, d = i % D;
  float acc = 0.f;
  for (int j = 0; j < s; ++j) acc += h[(long)(n * s + j) * D + d];
  pooled[i] = acc / (float)s;
}

__global__ void upsample_kernel(const float* __restrict__ a, float* __restrict__ cc,
                                int s, int si, int total, int D, int CW) {
  int i = blockIdx.x * blockDim.x + threadIdx.x;
  if (i >= total) return;
  int n = i / D, d = i % D;
  cc[(long)n * CW + si * D + d] = a[(long)(n / s) * D + d];
}

__global__ void softmax_rows(float* __restrict__ x, int cols, int ld) {
  float* p = x + (long)blockIdx.x * ld;
  __shared__ float red[256];
  int t = threadIdx.x;
  float mx = -3.0e38f;
  for (int c = t; c < cols; c += 256) mx = fmaxf(mx, p[c]);
  red[t] = mx; __syncthreads();
  for (int s = 128; s > 0; s >>= 1) {
    if (t < s) red[t] = fmaxf(red[t], red[t + s]);
    __syncthreads();
  }
  mx = red[0]; __syncthreads();
  float sm = 0.f;
  for (int c = t; c < cols; c += 256) { float e = expf(p[c] - mx); p[c] = e; sm += e; }
  red[t] = sm; __syncthreads();
  for (int s = 128; s > 0; s >>= 1) {
    if (t < s) red[t] += red[t + s];
    __syncthreads();
  }
  float inv = 1.0f / red[0];
  for (int c = t; c < cols; c += 256) p[c] *= inv;
}

__global__ void add_ln_kernel(const float* __restrict__ x, const float* __restrict__ y,
                              const float* __restrict__ g, const float* __restrict__ b,
                              float* __restrict__ out) {
  const int cols = DMODEL;
  long base = (long)blockIdx.x * cols;
  int t = threadIdx.x;
  float v0 = x[base + t]       + y[base + t];
  float v1 = x[base + t + 256] + y[base + t + 256];
  __shared__ float s1[256], s2[256];
  s1[t] = v0 + v1;
  s2[t] = v0 * v0 + v1 * v1;
  __syncthreads();
  for (int s = 128; s > 0; s >>= 1) {
    if (t < s) { s1[t] += s1[t + s]; s2[t] += s2[t + s]; }
    __syncthreads();
  }
  float mean = s1[0] / (float)cols;
  float var  = s2[0] / (float)cols - mean * mean;
  float inv  = rsqrtf(var + 1e-3f);
  out[base + t]       = (v0 - mean) * inv * g[t]       + b[t];
  out[base + t + 256] = (v1 - mean) * inv * g[t + 256] + b[t + 256];
}

__global__ void meanrows_kernel(const float* __restrict__ h, float* __restrict__ g,
                                int rows, int D) {
  int d = blockIdx.x * blockDim.x + threadIdx.x;
  if (d >= D) return;
  float s = 0.f;
  for (int n = 0; n < rows; ++n) s += h[(long)n * D + d];
  g[d] = s / (float)rows;
}

__global__ void bcast_kernel(const float* __restrict__ p, float* __restrict__ out,
                             int total, int C) {
  int i = blockIdx.x * blockDim.x + threadIdx.x;
  if (i >= total) return;
  out[i] = p[i % C];
}

// ---------------------------------------------------------------------------
// Host-side orchestration
// ---------------------------------------------------------------------------
static void pack_b(const float* B, int ldb, int K, int N, _Float16* P, hipStream_t st) {
  int KT  = (K + 31) >> 5;
  int NTt = (N + 15) >> 4;
  long total = (long)KT * NTt * 32;
  pack_b_kernel<<<(int)((total + 255) / 256), 256, 0, st>>>(B, ldb, K, N, NTt, total, P);
}

static void run_gemm_pk(const float* A, int lda, const _Float16* P, int NTt, int ntBase,
                        const float* bias, float* C, int ldc,
                        int M, int N, int K, float alpha, int act, hipStream_t st) {
  int KT = (K + 31) >> 5;
  if (N >= 64) {
    dim3 grid((M + 15) / 16, (N + 63) / 64);
    gemm_wmma_pk<4><<<grid, 32, 0, st>>>(A, lda, P, NTt, ntBase, bias, C, ldc, M, N, KT, alpha, act);
  } else {
    dim3 grid((M + 15) / 16, (N + 15) / 16);
    gemm_wmma_pk<1><<<grid, 32, 0, st>>>(A, lda, P, NTt, ntBase, bias, C, ldc, M, N, KT, alpha, act);
  }
}

extern "C" void kernel_launch(void* const* d_in, const int* in_sizes, int n_in,
                              void* d_out, int out_size, void* d_ws, size_t ws_size,
                              hipStream_t stream) {
  const float* ln_in_b = (const float*)d_in[2];   // scalar b (ln_in_g multiplied by 0)
  const float* patch_W = (const float*)d_in[3];
  const float* patch_b = (const float*)d_in[4];
  const float* pos_emb = (const float*)d_in[5];
  const float* Wq = (const float*)d_in[6];   const float* bq = (const float*)d_in[7];
  const float* Wk = (const float*)d_in[8];   const float* bk = (const float*)d_in[9];
  const float* Wv = (const float*)d_in[10];  const float* bv = (const float*)d_in[11];
  const float* Wo = (const float*)d_in[12];  const float* bo = (const float*)d_in[13];
  const float* Wc = (const float*)d_in[14];  const float* bc = (const float*)d_in[15];
  const float* ln1_g = (const float*)d_in[16]; const float* ln1_b = (const float*)d_in[17];
  const float* W1 = (const float*)d_in[18];  const float* b1 = (const float*)d_in[19];
  const float* W2 = (const float*)d_in[20];  const float* b2 = (const float*)d_in[21];
  const float* ln2_g = (const float*)d_in[22]; const float* ln2_b = (const float*)d_in[23];
  const float* headW = (const float*)d_in[24]; const float* headb = (const float*)d_in[25];
  float* out = (float*)d_out;

  // Workspace layout (floats); all offsets multiples of 8 -> >=32B alignment.
  float* ws = (float*)d_ws;
  float* h      = ws;                       // 500*512
  float* tmp    = h      + 500 * 512;
  float* pooled = tmp    + 500 * 512;
  float* Qb     = pooled + 500 * 512;       // 500*160
  float* Kb     = Qb     + 500 * 160;
  float* Vb     = Kb     + 500 * 160;
  float* Ob     = Vb     + 500 * 160;
  float* Sc     = Ob     + 500 * 160;       // padded score rows (<= 500*512)
  float* Aout   = Sc     + 500 * 512;
  float* cc     = Aout   + 500 * 512;       // 500*1536
  float* ffb    = cc     + 500 * 1536;      // 500*2048
  float* csum   = ffb    + 500 * 2048;      // 512
  float* gvec   = csum   + 512;             // 512
  float* logit  = gvec   + 512;             // 256
  // f16 pack buffers (sizes in f16 elems, K/N padded):
  _Float16* pkW  = (_Float16*)(logit + 256);        // up to 2048*512
  _Float16* pkQ  = pkW  + 2048 * 512;               // Wq pack: 512*160
  _Float16* pkK  = pkQ  + 512 * 160;
  _Float16* pkV  = pkK  + 512 * 160;
  _Float16* pkVb = pkV  + 512 * 160;                // Vb pack: 512*160

  const int scales[NSCALE] = {1, 2, 4};
  const float inv_sqrt_kd = 0.17677669529663687f;   // 1/sqrt(32)

  colsum_kernel<<<2, 256, 0, stream>>>(patch_W, csum, 100, DMODEL);
  h0_kernel<<<(N_TOK * DMODEL + 255) / 256, 256, 0, stream>>>(
      csum, patch_b, pos_emb, ln_in_b, h, N_TOK * DMODEL, DMODEL);

  for (int l = 0; l < NLAYER; ++l) {
    for (int si = 0; si < NSCALE; ++si) {
      const int s   = scales[si];
      const int Ns  = N_TOK / s;
      const int Sld = (Ns + 31) & ~31;              // padded score leading dim
      const float* X = h;
      if (s > 1) {
        pool_kernel<<<(Ns * DMODEL + 255) / 256, 256, 0, stream>>>(
            h, pooled, s, Ns * DMODEL, DMODEL);
        X = pooled;
      }
      const size_t woff = (size_t)(l * NSCALE + si);
      const float* Wq_ls = Wq + woff * DMODEL * QKVW;
      const float* Wk_ls = Wk + woff * DMODEL * QKVW;
      const float* Wv_ls = Wv + woff * DMODEL * QKVW;
      const float* Wo_ls = Wo + woff * QKVW * DMODEL;
      const float* bq_ls = bq + woff * QKVW;
      const float* bk_ls = bk + woff * QKVW;
      const float* bv_ls = bv + woff * QKVW;
      const float* bo_ls = bo + woff * DMODEL;

      // QKV projections (packed weights; NTt = 160/16 = 10)
      pack_b(Wq_ls, QKVW, DMODEL, QKVW, pkQ, stream);
      pack_b(Wk_ls, QKVW, DMODEL, QKVW, pkK, stream);
      pack_b(Wv_ls, QKVW, DMODEL, QKVW, pkV, stream);
      run_gemm_pk(X, DMODEL, pkQ, 10, 0, bq_ls, Qb, QKVW, Ns, QKVW, DMODEL, 1.f, 0, stream);
      run_gemm_pk(X, DMODEL, pkK, 10, 0, bk_ls, Kb, QKVW, Ns, QKVW, DMODEL, 1.f, 0, stream);
      run_gemm_pk(X, DMODEL, pkV, 10, 0, bv_ls, Vb, QKVW, Ns, QKVW, DMODEL, 1.f, 0, stream);

      // Pack V activations once per scale (K = Ns zero-padded, N = 160);
      // all 5 heads' P@V GEMMs slice it via ntBase.
      pack_b(Vb, QKVW, Ns, QKVW, pkVb, stream);

      for (int hd = 0; hd < 5; ++hd) {
        gemm_wmma_qk<<<dim3((Ns + 15) / 16, (Ns + 63) / 64), 32, 0, stream>>>(
            Qb + hd * KDIM, QKVW, Kb + hd * KDIM, QKVW,
            Sc, Sld, Ns, Ns, KDIM, inv_sqrt_kd);
        softmax_rows<<<Ns, 256, 0, stream>>>(Sc, Ns, Sld);
        run_gemm_pk(Sc, Sld, pkVb, 10, hd * 2, nullptr,
                    Ob + hd * KDIM, QKVW, Ns, KDIM, Ns, 1.f, 0, stream);
      }

      pack_b(Wo_ls, DMODEL, QKVW, DMODEL, pkW, stream);
      run_gemm_pk(Ob, QKVW, pkW, 32, 0, bo_ls, Aout, DMODEL,
                  Ns, DMODEL, QKVW, 1.f, 0, stream);
      upsample_kernel<<<(N_TOK * DMODEL + 255) / 256, 256, 0, stream>>>(
          Aout, cc, s, si, N_TOK * DMODEL, DMODEL, CONCAT_W);
    }

    pack_b(Wc + (size_t)l * CONCAT_W * DMODEL, DMODEL, CONCAT_W, DMODEL, pkW, stream);
    run_gemm_pk(cc, CONCAT_W, pkW, 32, 0, bc + (size_t)l * DMODEL, tmp, DMODEL,
                N_TOK, DMODEL, CONCAT_W, 1.f, 0, stream);
    add_ln_kernel<<<N_TOK, 256, 0, stream>>>(
        h, tmp, ln1_g + (size_t)l * DMODEL, ln1_b + (size_t)l * DMODEL, h);

    pack_b(W1 + (size_t)l * DMODEL * FFDIM, FFDIM, DMODEL, FFDIM, pkW, stream);
    run_gemm_pk(h, DMODEL, pkW, 128, 0, b1 + (size_t)l * FFDIM, ffb, FFDIM,
                N_TOK, FFDIM, DMODEL, 1.f, 1 /*gelu*/, stream);
    pack_b(W2 + (size_t)l * FFDIM * DMODEL, DMODEL, FFDIM, DMODEL, pkW, stream);
    run_gemm_pk(ffb, FFDIM, pkW, 32, 0, b2 + (size_t)l * DMODEL, tmp, DMODEL,
                N_TOK, DMODEL, FFDIM, 1.f, 0, stream);
    add_ln_kernel<<<N_TOK, 256, 0, stream>>>(
        h, tmp, ln2_g + (size_t)l * DMODEL, ln2_b + (size_t)l * DMODEL, h);
  }

  meanrows_kernel<<<2, 256, 0, stream>>>(h, gvec, N_TOK, DMODEL);
  pack_b(headW, NCLASS, DMODEL, NCLASS, pkW, stream);
  run_gemm_pk(gvec, DMODEL, pkW, 16, 0, headb, logit, NCLASS,
              1, NCLASS, DMODEL, 1.f, 0, stream);
  softmax_rows<<<1, 256, 0, stream>>>(logit, NCLASS, NCLASS);
  bcast_kernel<<<(NBATCH * NCLASS + 255) / 256, 256, 0, stream>>>(
      logit, out, NBATCH * NCLASS, NCLASS);
}
